// T5Attention_11639361372632
// MI455X (gfx1250) — compile-verified, matrix operand order checked by reading, not run on hardware
//
#include <hip/hip_runtime.h>
#include <hip/hip_bf16.h>

// ---------------------------------------------------------------------------
// T5 attention for MI455X (gfx1250, wave32, WMMA).
//  B=2, S=2048, D=1024, H=16, dh=64.
//  Pipeline: 3x bf16 WMMA GEMM (QKV) -> flash-attention WMMA kernel.
// Compute-bound (~60 GFLOP vs ~50 MB traffic), so everything runs on
// v_wmma_f32_16x16x32_bf16 with fp32 accumulation; scores are never
// materialized (online softmax).
// ---------------------------------------------------------------------------

#define S_LEN   2048
#define D_MODEL 1024
#define NBATCH  2
#define NHEADS  16
#define DHEAD   64
#define MTOT    (NBATCH * S_LEN)   // 4096 rows for the projections

typedef __attribute__((ext_vector_type(16))) __bf16 v16bf;
typedef __attribute__((ext_vector_type(8)))  __bf16 v8bf;
typedef __attribute__((ext_vector_type(8)))  float  v8f;
typedef __attribute__((ext_vector_type(4)))  int    v4i;

// gfx1250 async global->LDS path (ASYNCcnt). Guarded: falls back to a
// synchronous copy if this toolchain doesn't declare the builtins.
#if defined(__AMDGCN__) && defined(__has_builtin)
#if __has_builtin(__builtin_amdgcn_global_load_async_to_lds_b128) && \
    __has_builtin(__builtin_amdgcn_s_wait_asynccnt)
#define USE_ASYNC_LDS 1
#endif
#endif
#ifndef USE_ASYNC_LDS
#define USE_ASYNC_LDS 0
#endif

#if USE_ASYNC_LDS
typedef __attribute__((address_space(1))) v4i GV4i;   // global b128 payload
typedef __attribute__((address_space(3))) v4i LV4i;   // LDS b128 payload
#endif

// Native converts (v_cvt on gfx1250) instead of hand-rolled RNE bit math.
static __device__ __forceinline__ __bf16 f2bf(float f) { return (__bf16)f; }

static __device__ __forceinline__ v8f wmma_bf16(v16bf a, v16bf b, v8f c) {
  // D = A(16x32 bf16) * B(32x16 bf16) + C(16x16 f32)
  return __builtin_amdgcn_wmma_f32_16x16x32_bf16(false, a, false, b,
                                                 (short)0, c, false, false);
}

// ---------------------------------------------------------------------------
// Kernel 1: C = X*W + bias, fp32 in, bf16 out, reshaped to [B,H,S,dh].
// 128x128 tile / workgroup, 8 waves (2 M-slabs x 4 N-slabs), K-step 32.
// ---------------------------------------------------------------------------
__global__ __launch_bounds__(256)
void qkv_gemm_kernel(const float* __restrict__ X,   // [MTOT, D]
                     const float* __restrict__ W,   // [D, D]
                     const float* __restrict__ bias,// [D]
                     __bf16* __restrict__ out) {    // [B,H,S,dh] bf16
  __shared__ __attribute__((aligned(64))) __bf16 Xt[128 * 32];  // [m][k]
  __shared__ __attribute__((aligned(64))) __bf16 Wt[128 * 32];  // [n][k] (transposed)

  const int tid  = threadIdx.x;
  const int lane = tid & 31;
  const int wave = tid >> 5;
  const int wm   = wave >> 2;          // 0..1  -> 64-row slab
  const int wn   = wave & 3;           // 0..3  -> 32-col slab
  const int lh   = lane >> 4;          // half-wave select
  const int ln   = lane & 15;
  const int mbase = blockIdx.y * 128;
  const int nbase = blockIdx.x * 128;

  v8f acc[4][2];
#pragma unroll
  for (int mi = 0; mi < 4; ++mi)
#pragma unroll
    for (int ni = 0; ni < 2; ++ni)
#pragma unroll
      for (int i = 0; i < 8; ++i) acc[mi][ni][i] = 0.0f;

  for (int k0 = 0; k0 < D_MODEL; k0 += 32) {
    // Stage X tile (128x32 fp32 -> bf16); one contiguous v16bf store/thread.
    {
      const int row  = tid >> 1;
      const int koff = (tid & 1) * 16;
      const float* src = X + (size_t)(mbase + row) * D_MODEL + k0 + koff;
      v16bf xv;
#pragma unroll
      for (int i = 0; i < 16; i += 4) {
        float4 v = *(const float4*)(src + i);
        xv[i + 0] = f2bf(v.x); xv[i + 1] = f2bf(v.y);
        xv[i + 2] = f2bf(v.z); xv[i + 3] = f2bf(v.w);
      }
      *(v16bf*)&Xt[row * 32 + koff] = xv;
    }
    // Stage W tile transposed: thread owns one n-column (reads coalesced
    // across lanes along n; one contiguous v16bf LDS store along k).
    {
      const int n  = tid & 127;
      const int kh = (tid >> 7) * 16;
      const float* src = W + (size_t)(k0 + kh) * D_MODEL + nbase + n;
      v16bf wv;
#pragma unroll
      for (int i = 0; i < 16; ++i)
        wv[i] = f2bf(src[(size_t)i * D_MODEL]);
      *(v16bf*)&Wt[n * 32 + kh] = wv;
    }
    __syncthreads();

    v16bf bfrag[2];
#pragma unroll
    for (int ni = 0; ni < 2; ++ni) {
      const int ncol = wn * 32 + ni * 16 + ln;
      bfrag[ni] = *(const v16bf*)&Wt[ncol * 32 + lh * 16];
    }
#pragma unroll
    for (int mi = 0; mi < 4; ++mi) {
      const int row = wm * 64 + mi * 16 + ln;
      v16bf afrag = *(const v16bf*)&Xt[row * 32 + lh * 16];
      acc[mi][0] = wmma_bf16(afrag, bfrag[0], acc[mi][0]);
      acc[mi][1] = wmma_bf16(afrag, bfrag[1], acc[mi][1]);
    }
    __syncthreads();
  }

  // Epilogue: + bias, cast to bf16, scatter into [B,H,S,dh].
#pragma unroll
  for (int ni = 0; ni < 2; ++ni) {
    const int Ni = nbase + wn * 32 + ni * 16 + ln;
    const float bv = bias[Ni];
    const int h = Ni >> 6, f = Ni & 63;
#pragma unroll
    for (int mi = 0; mi < 4; ++mi) {
#pragma unroll
      for (int r = 0; r < 8; ++r) {
        const int Mi = mbase + wm * 64 + mi * 16 + r + (lh << 3);
        const int b = Mi >> 11, s = Mi & (S_LEN - 1);
        out[(((size_t)(b * NHEADS + h) * S_LEN + s) * DHEAD) + f] =
            f2bf(acc[mi][ni][r] + bv);
      }
    }
  }
}

// ---------------------------------------------------------------------------
// Kernel 2: flash attention. Grid (S/128, B*H). 8 waves; each wave owns 16
// query rows (16x64 fp32 accumulator = 4 C-tiles), streams 32-key tiles.
// ---------------------------------------------------------------------------
__global__ __launch_bounds__(256)
void t5_attention_kernel(const __bf16* __restrict__ Q,   // [B,H,S,dh]
                         const __bf16* __restrict__ K,   // [B,H,S,dh]
                         const __bf16* __restrict__ V,   // [B,H,S,dh]
                         const float*  __restrict__ rel_bias, // [32,16]
                         float* __restrict__ out) {      // [B,S,D] fp32
  __shared__ __attribute__((aligned(64))) __bf16 Klds[32 * DHEAD];  // [key][feat]
  __shared__ __attribute__((aligned(64))) __bf16 Vt[DHEAD * 32];    // [feat][key]
  __shared__ __attribute__((aligned(64))) __bf16 Ps[8][16 * 32];    // per-wave P
  __shared__ float rbs[32];                 // this head's bias buckets

  const int tid  = threadIdx.x;
  const int lane = tid & 31;
  const int wave = tid >> 5;
  const int lh   = lane >> 4;
  const int ln   = lane & 15;
  const int bh   = blockIdx.y;            // b*H + h
  const int b    = bh >> 4, h = bh & 15;
  const int qbase = blockIdx.x * 128 + wave * 16;

  const __bf16* Qh = Q + (size_t)bh * S_LEN * DHEAD;
  const __bf16* Kh = K + (size_t)bh * S_LEN * DHEAD;
  const __bf16* Vh = V + (size_t)bh * S_LEN * DHEAD;

  if (tid < 32) rbs[tid] = rel_bias[tid * NHEADS + h];

  // Q fragments for this wave's 16 rows (dh=64 -> two K=32 chunks).
  v16bf qa[2];
#pragma unroll
  for (int c = 0; c < 2; ++c)
    qa[c] = *(const v16bf*)&Qh[(size_t)(qbase + ln) * DHEAD + c * 32 + lh * 16];

  v8f o[4];
#pragma unroll
  for (int f = 0; f < 4; ++f)
#pragma unroll
    for (int i = 0; i < 8; ++i) o[f][i] = 0.0f;
  float m_run[8], l_run[8];
#pragma unroll
  for (int r = 0; r < 8; ++r) { m_run[r] = -1e30f; l_run[r] = 0.0f; }

  const float scale = 0.125f;   // 1/sqrt(dh)
  __syncthreads();              // rbs visible

  for (int j = 0; j < S_LEN; j += 32) {
    // --- stage K tile [32x64] row-major, V tile transposed [64x32] ---
    {
      const int e = tid * 8;                 // 2048 bf16 / 256 threads
#if USE_ASYNC_LDS
      // ASYNCcnt-tracked DMA into LDS, no VGPR round-trip.
      __builtin_amdgcn_global_load_async_to_lds_b128(
          (GV4i*)(Kh + (size_t)j * DHEAD + e), (LV4i*)&Klds[e], 0, 0);
#else
      *(v8bf*)&Klds[e] = *(const v8bf*)&Kh[(size_t)j * DHEAD + e];
#endif
      v8bf vv = *(const v8bf*)&Vh[(size_t)j * DHEAD + e];
      const int key = e >> 6, feat = e & 63;
#pragma unroll
      for (int i = 0; i < 8; ++i) Vt[(feat + i) * 32 + key] = vv[i];
      if (j + 32 < S_LEN) {                  // global_prefetch_b8 for next tile
        __builtin_prefetch(&Kh[(size_t)(j + 32) * DHEAD + e], 0, 1);
        __builtin_prefetch(&Vh[(size_t)(j + 32) * DHEAD + e], 0, 1);
      }
    }
#if USE_ASYNC_LDS
    __builtin_amdgcn_s_wait_asynccnt(0);
#endif
    __syncthreads();

    // --- scores: S = Q (16x64) * K^T (64x32), two 16x16 C-tiles ---
    v8f s0, s1;
#pragma unroll
    for (int i = 0; i < 8; ++i) { s0[i] = 0.0f; s1[i] = 0.0f; }
#pragma unroll
    for (int c = 0; c < 2; ++c) {
      v16bf kb0 = *(const v16bf*)&Klds[(ln) * DHEAD + c * 32 + lh * 16];
      v16bf kb1 = *(const v16bf*)&Klds[(16 + ln) * DHEAD + c * 32 + lh * 16];
      s0 = wmma_bf16(qa[c], kb0, s0);
      s1 = wmma_bf16(qa[c], kb1, s1);
    }

    // --- T5 bucket bias (LDS table) + online softmax ---
    float p0[8], p1[8], alpha[8];
#pragma unroll
    for (int r = 0; r < 8; ++r) {
      const int ctx = qbase + r + (lh << 3);
      int n0 = ctx - (j + ln);       n0 = n0 > 0 ? n0 : 0;
      int n1 = ctx - (j + 16 + ln);  n1 = n1 > 0 ? n1 : 0;
      int b0, b1;
      if (n0 < 16) b0 = n0;
      else { int t0 = 16 + (int)(__log2f((float)n0 * 0.0625f) * 16.0f); b0 = t0 < 31 ? t0 : 31; }
      if (n1 < 16) b1 = n1;
      else { int t1 = 16 + (int)(__log2f((float)n1 * 0.0625f) * 16.0f); b1 = t1 < 31 ? t1 : 31; }
      const float v0 = s0[r] * scale + rbs[b0];
      const float v1 = s1[r] * scale + rbs[b1];
      float t = fmaxf(v0, v1);
#pragma unroll
      for (int d = 1; d < 16; d <<= 1) t = fmaxf(t, __shfl_xor(t, d));
      const float mnew = fmaxf(m_run[r], t);
      const float a = __expf(m_run[r] - mnew);
      const float e0 = __expf(v0 - mnew);
      const float e1 = __expf(v1 - mnew);
      float rs = e0 + e1;
#pragma unroll
      for (int d = 1; d < 16; d <<= 1) rs += __shfl_xor(rs, d);
      l_run[r] = l_run[r] * a + rs;
      m_run[r] = mnew;
      alpha[r] = a;
      p0[r] = e0; p1[r] = e1;
    }
#pragma unroll
    for (int f = 0; f < 4; ++f)
#pragma unroll
      for (int r = 0; r < 8; ++r) o[f][r] *= alpha[r];

    // --- repack P (C layout) -> A fragment via wave-private LDS ---
    __bf16* ps = &Ps[wave][0];
#pragma unroll
    for (int r = 0; r < 8; ++r) {
      const int row = r + (lh << 3);
      ps[row * 32 + ln]      = f2bf(p0[r]);
      ps[row * 32 + 16 + ln] = f2bf(p1[r]);
    }
    v16bf pa = *(const v16bf*)&ps[ln * 32 + lh * 16];

    // --- O += P (16x32) * V (32x64): 4 WMMAs over feature tiles ---
#pragma unroll
    for (int f = 0; f < 4; ++f) {
      v16bf vb = *(const v16bf*)&Vt[(f * 16 + ln) * 32 + lh * 16];
      o[f] = wmma_bf16(pa, vb, o[f]);
    }
    __syncthreads();
  }

  // --- normalize and store fp32 context [B,S,D] ---
#pragma unroll
  for (int f = 0; f < 4; ++f) {
#pragma unroll
    for (int r = 0; r < 8; ++r) {
      const int qrow = qbase + r + (lh << 3);
      const int feat = f * 16 + ln;
      out[(size_t)(b * S_LEN + qrow) * D_MODEL + h * DHEAD + feat] =
          o[f][r] / l_run[r];
    }
  }
}

// ---------------------------------------------------------------------------
extern "C" void kernel_launch(void* const* d_in, const int* in_sizes, int n_in,
                              void* d_out, int out_size, void* d_ws, size_t ws_size,
                              hipStream_t stream) {
  const float* X  = (const float*)d_in[0];  // hidden_states [B,S,D]
  const float* Wq = (const float*)d_in[1];
  const float* bq = (const float*)d_in[2];
  const float* Wk = (const float*)d_in[3];
  const float* bk = (const float*)d_in[4];
  const float* Wv = (const float*)d_in[5];
  const float* bv = (const float*)d_in[6];
  const float* rb = (const float*)d_in[7];  // rel_bias [32,16]
  float* out = (float*)d_out;

  const size_t per = (size_t)MTOT * D_MODEL;       // bf16 elements per tensor
  unsigned short* ws = (unsigned short*)d_ws;      // 3 x 8 MB bf16 workspace
  __bf16* qws = (__bf16*)(ws);
  __bf16* kws = (__bf16*)(ws + per);
  __bf16* vws = (__bf16*)(ws + 2 * per);

  dim3 ggrid(D_MODEL / 128, MTOT / 128, 1);
  qkv_gemm_kernel<<<ggrid, 256, 0, stream>>>(X, Wq, bq, qws);
  qkv_gemm_kernel<<<ggrid, 256, 0, stream>>>(X, Wk, bk, kws);
  qkv_gemm_kernel<<<ggrid, 256, 0, stream>>>(X, Wv, bv, vws);

  dim3 agrid(S_LEN / 128, NBATCH * NHEADS, 1);
  t5_attention_kernel<<<agrid, 256, 0, stream>>>(qws, kws, vws, rb, out);
}
